// KernelPoolingLayer_80616536146799
// MI455X (gfx1250) — compile-verified, matrix operand order checked by reading, not run on hardware
//
#include <hip/hip_runtime.h>
#include <hip/hip_bf16.h>

typedef __attribute__((ext_vector_type(16))) _Float16 v16h;
typedef __attribute__((ext_vector_type(8)))  float    v8f;

#define L_   2
#define B_   64
#define Q_   64
#define D_   1024
#define K_   11
#define QT_  4      // Q / 16
#define REC_ 176    // K_ * 16 floats per workspace record

// ---------------------------------------------------------------------------
// Kernel 1: per-(l,b,qtile,dslice) wave computes sum over its d-range of
// exp(-(x-mu_k)^2/(2 sigma_k^2)) * doc_mask, for 16 q values x 11 kernels,
// using V_WMMA_F32_16X16X32_F16 with an all-ones A matrix as the reducer.
//   B-matrix (32x16 f16): lane n<16 holds column n (q = qt*16+n), K=d 0..15
//                         lane n+16 holds column n,              K=d 16..31
//   -> each lane loads 16 consecutive fp32 d-values (4x b128, fully
//      utilizing 128B lines across the two half-waves), lane-local math.
//   D readout: lane n<16, acc element 0 = D[M=0, N=n] = column sum.
// exp argument for k>=1 evaluated as c*x^2 + b*x + cc (x^2 shared across all
// kernels -> 2 fma per (element,kernel)); k=0 (sigma=1e-3, |c|~7e5) keeps the
// cancellation-free direct (x-mu)^2 form.
// ---------------------------------------------------------------------------
__global__ __launch_bounds__(32)
void knrm_pool_wmma(const float* __restrict__ mm, const float* __restrict__ qdm,
                    float* __restrict__ ws, int dsplit, int drange) {
  const float mu[K_] = {1.0f, 0.9f, 0.7f, 0.5f, 0.3f, 0.1f,
                        -0.1f, -0.3f, -0.5f, -0.7f, -0.9f};
  // c2_k = -(1/(2*sigma_k^2)) * log2(e): sigma_0 = 1e-3, sigma_k>0 = 0.1
  const float c2[K_] = {-721347.5204444817f,
                        -72.13475204444817f, -72.13475204444817f,
                        -72.13475204444817f, -72.13475204444817f,
                        -72.13475204444817f, -72.13475204444817f,
                        -72.13475204444817f, -72.13475204444817f,
                        -72.13475204444817f, -72.13475204444817f};

  int wid = blockIdx.x;
  int ds  = wid % dsplit;
  int qt  = (wid / dsplit) & (QT_ - 1);
  int lb  = wid / (dsplit * QT_);          // l*B + b
  int b   = lb & (B_ - 1);

  int lane = threadIdx.x;
  int n    = lane & 15;                    // q within tile / N column
  int hi   = lane >> 4;                    // 0: d 0..15, 1: d 16..31 of chunk

  size_t q    = (size_t)qt * 16 + n;
  size_t base = (size_t)ds * drange + (size_t)hi * 16;
  const float4* mp = (const float4*)(mm  + ((size_t)lb * Q_ + q) * D_ + base);
  const float4* kp = (const float4*)(qdm + ((size_t)b  * Q_ + q) * D_ + base);

  v16h ones;
#pragma unroll
  for (int i = 0; i < 16; ++i) ones[i] = (_Float16)1.0f;

  v8f acc[K_];
#pragma unroll
  for (int k = 0; k < K_; ++k) acc[k] = (v8f){0.f,0.f,0.f,0.f,0.f,0.f,0.f,0.f};

  int chunks = drange >> 5;                // 32 d per WMMA step
  for (int c = 0; c < chunks; ++c) {
    float x[16], w[16], x2[16];
#pragma unroll
    for (int j = 0; j < 4; ++j) {
      float4 xv = mp[c * 8 + j];           // chunk advance = 32 floats = 8xf4
      float4 wv = kp[c * 8 + j];
      x[j*4+0] = xv.x; x[j*4+1] = xv.y; x[j*4+2] = xv.z; x[j*4+3] = xv.w;
      w[j*4+0] = wv.x; w[j*4+1] = wv.y; w[j*4+2] = wv.z; w[j*4+3] = wv.w;
    }
#pragma unroll
    for (int i = 0; i < 16; ++i) x2[i] = x[i] * x[i];

#pragma unroll
    for (int k = 0; k < K_; ++k) {
      v16h bm;
      if (k == 0) {
        // sigma = 1e-3: keep exact (x-mu)^2 to avoid catastrophic cancellation
#pragma unroll
        for (int i = 0; i < 16; ++i) {
          float d = x[i] - mu[0];
          float e = __builtin_amdgcn_exp2f(d * d * c2[0]);
          bm[i] = (_Float16)(e * w[i]);
        }
      } else {
        const float bk = -2.0f * c2[k] * mu[k];        // folded at compile time
        const float ck = c2[k] * mu[k] * mu[k];
#pragma unroll
        for (int i = 0; i < 16; ++i) {
          float arg = fmaf(c2[k], x2[i], fmaf(bk, x[i], ck));
          float e   = __builtin_amdgcn_exp2f(arg);
          bm[i] = (_Float16)(e * w[i]);
        }
      }
      // 8 args: (neg_a, A, neg_b, B, c_mod, C, reuse_a, reuse_b)
      acc[k] = __builtin_amdgcn_wmma_f32_16x16x32_f16(
          false, ones, false, bm, (short)0, acc[k], false, false);
    }
  }

  if (lane < 16) {
    float* op = ws + ((size_t)(lb * QT_ + qt) * dsplit + ds) * REC_ + n;
#pragma unroll
    for (int k = 0; k < K_; ++k) op[k * 16] = acc[k][0];
  }
}

// ---------------------------------------------------------------------------
// Kernel 2: combine d-split partials, log/clip/scale/pad-mask, reduce over Q.
// One block per (l,b); 64 threads = one q each.
// ---------------------------------------------------------------------------
__global__ __launch_bounds__(64)
void knrm_pool_reduce(const float* __restrict__ ws, const float* __restrict__ qpad,
                      float* __restrict__ out, int dsplit) {
  __shared__ float sm[K_ * Q_];
  int lb = blockIdx.x;
  int b  = lb & (B_ - 1);
  int qv = threadIdx.x;                    // 0..63
  int qt = qv >> 4, n = qv & 15;
  float pad = qpad[(size_t)b * Q_ + qv];

#pragma unroll
  for (int k = 0; k < K_; ++k) {
    float s = 0.f;
    for (int ds = 0; ds < dsplit; ++ds)
      s += ws[((size_t)(lb * QT_ + qt) * dsplit + ds) * REC_ + k * 16 + n];
    sm[k * Q_ + qv] = 0.01f * __logf(fmaxf(s, 1e-10f)) * pad;
  }
  __syncthreads();

  if (qv < K_) {
    float a = 0.f;
#pragma unroll 8
    for (int i = 0; i < Q_; ++i) a += sm[qv * Q_ + i];
    out[(size_t)lb * K_ + qv] = a;
  }
}

// ---------------------------------------------------------------------------
extern "C" void kernel_launch(void* const* d_in, const int* in_sizes, int n_in,
                              void* d_out, int out_size, void* d_ws, size_t ws_size,
                              hipStream_t stream) {
  const float* mm  = (const float*)d_in[0];   // [L,B,Q,D] f32
  const float* qdm = (const float*)d_in[1];   // [B,Q,D]   f32
  const float* qpd = (const float*)d_in[2];   // [B,Q]     f32
  float* out = (float*)d_out;                 // [L,B,K]   f32
  float* ws  = (float*)d_ws;

  int dsplit = 8;                             // D-axis parallelism (4096 waves)
  while (dsplit > 1 &&
         (size_t)L_ * B_ * QT_ * dsplit * REC_ * sizeof(float) > ws_size)
    dsplit >>= 1;
  int drange = D_ / dsplit;

  dim3 g1(L_ * B_ * QT_ * dsplit), t1(32);
  knrm_pool_wmma<<<g1, t1, 0, stream>>>(mm, qdm, ws, dsplit, drange);

  dim3 g2(L_ * B_), t2(Q_);
  knrm_pool_reduce<<<g2, t2, 0, stream>>>(ws, qpd, out, dsplit);
}
